// Transformer_77000173682975
// MI455X (gfx1250) — matrix operand order repaired and checked by reading, not compiled, not run on hardware
//
#include <hip/hip_runtime.h>

// ---------------------------------------------------------------------------
// MI455X (gfx1250, wave32) transformer forward pass.
// GEMMs + attention matmuls: V_WMMA_F32_16X16X32_BF16.
// GEMM tile staging: Tensor Data Mover (tensor_load_to_lds + s_wait_tensorcnt)
// with double-buffered LDS, falling back to vector loads if the builtin is
// not available on this toolchain.
// ---------------------------------------------------------------------------

typedef __attribute__((ext_vector_type(16))) __bf16 v16bf;
typedef __attribute__((ext_vector_type(8)))  float  v8f;

#if defined(__gfx1250__) && __has_builtin(__builtin_amdgcn_tensor_load_to_lds)
#define USE_TDM 1
#else
#define USE_TDM 0
#endif

__device__ __forceinline__ v8f wmma_bf16(v16bf a, v16bf b, v8f c) {
  return __builtin_amdgcn_wmma_f32_16x16x32_bf16(false, a, false, b, (short)0, c,
                                                 false, false);
}

__device__ __forceinline__ float gelu_f(float x) {
  const float c = 0.7978845608028654f;  // sqrt(2/pi)
  float x3 = x * x * x;
  return 0.5f * x * (1.0f + tanhf(c * (x + 0.044715f * x3)));
}

constexpr int BM = 128, BN = 128, BKg = 32, PAD = 8;

#if USE_TDM
typedef unsigned int v4u  __attribute__((ext_vector_type(4)));
typedef int          tv4i __attribute__((ext_vector_type(4)));
typedef int          tv8i __attribute__((ext_vector_type(8)));

// Issue one TDM 2-D tile load: rows x tile_k bf16 elements, row stride
// stride_k elements, into LDS at lds_addr with hardware padding of 4 DWORDs
// (8 bf16) after every 16 DWORDs (one 64B row) -> matches [rows][tile_k+PAD].
__device__ __forceinline__ void tdm_load_tile_bf16(unsigned lds_addr,
                                                   const __bf16* gptr,
                                                   unsigned tensor_k,
                                                   unsigned stride_k,
                                                   unsigned rows,
                                                   unsigned tile_k) {
  const unsigned long long ga = (unsigned long long)(uintptr_t)gptr;
  // D# group 0: count=1 | lds_addr | global_addr[56:0] | type=2
  v4u g0 = {1u, lds_addr, (unsigned)ga,
            (unsigned)((ga >> 32) & 0x01ffffffull) | (2u << 30)};
  // D# group 1: data_size=2B, pad_enable, pad_interval=16dw(code3),
  //             pad_amount=4dw(code3); dims/strides per ISA bit layout.
  const unsigned w0 = (1u << 16) | (1u << 20) | (3u << 22) | (3u << 25);
  tv8i g1 = {(int)w0,
             (int)((tensor_k & 0xffffu) << 16),                        // dim0 lo16
             (int)(((tensor_k >> 16) & 0xffffu) | ((rows & 0xffffu) << 16)),
             (int)(((rows >> 16) & 0xffffu) | ((tile_k & 0xffffu) << 16)),
             (int)(rows & 0xffffu),                                    // tile_dim1
             (int)stride_k,                                            // dim0 stride
             0, 0};
  tv4i z4 = {0, 0, 0, 0};
  tv8i z8 = {0, 0, 0, 0, 0, 0, 0, 0};
  // amdgpu-toolchain (clang-23) 6-arg form: (g0, g1, g2, g3, g4, cpol)
  __builtin_amdgcn_tensor_load_to_lds(g0, g1, z4, z4, z8, 0);
}
#endif

// ------------------------------- converts ---------------------------------

__global__ __launch_bounds__(256) void f32_to_bf16_k(const float* __restrict__ in,
                                                     __bf16* __restrict__ out,
                                                     long long n) {
  long long i = (long long)blockIdx.x * blockDim.x + threadIdx.x;
  long long stride = (long long)gridDim.x * blockDim.x;
  for (; i < n; i += stride) out[i] = (__bf16)in[i];
}

// in: [L][R][C] f32 ; out: [L][C][R] bf16 (layer = blockIdx.y)
__global__ __launch_bounds__(256) void transpose_to_bf16_k(const float* __restrict__ in,
                                                           __bf16* __restrict__ out,
                                                           int R, int C) {
  const size_t layer = blockIdx.y;
  const float* src = in  + layer * (size_t)R * C;
  __bf16*      dst = out + layer * (size_t)R * C;
  int idx = blockIdx.x * 256 + threadIdx.x;
  if (idx < R * C) {
    int r = idx / C, c = idx % C;
    dst[(size_t)c * R + r] = (__bf16)src[idx];
  }
}

// --------------------------------- GEMM -----------------------------------
// C(MxN) = A(MxK, bf16 row-major) * Bt(NxK, bf16 row-major)^T
// EPI 0: f32 out. EPI 1: +bias, GELU, bf16 out. EPI 2: +bias, f32 out.
template <int EPI>
__global__ __launch_bounds__(256) void gemm_bf16_wmma(
    const __bf16* __restrict__ A, const __bf16* __restrict__ Bt,
    const float* __restrict__ bias, float* __restrict__ Cf,
    __bf16* __restrict__ Cb, int M, int N, int K) {
  __shared__ __bf16 As[2][BM][BKg + PAD];
  __shared__ __bf16 Bs[2][BN][BKg + PAD];

  const int tid  = threadIdx.x;
  const int lane = tid & 31;
  const int wave = tid >> 5;
  const int half = lane >> 4;
  const int r    = lane & 15;
  const int wm   = wave & 3;   // 4 M-strips of 32 rows
  const int wn   = wave >> 2;  // 2 N-strips of 64 cols
  const int m0   = blockIdx.x * BM;
  const int n0   = blockIdx.y * BN;

  const v8f vzero = {0.f, 0.f, 0.f, 0.f, 0.f, 0.f, 0.f, 0.f};
  v8f acc[2][4];
#pragma unroll
  for (int i = 0; i < 2; ++i)
#pragma unroll
    for (int j = 0; j < 4; ++j) acc[i][j] = vzero;

  const int nk = K / BKg;

#if USE_TDM
  if (wave == 0) {  // wave0 drives the Tensor Data Mover for the block
    tdm_load_tile_bf16((unsigned)(uintptr_t)&As[0][0][0],
                       A + (size_t)m0 * K, (unsigned)K, (unsigned)K, BM, BKg);
    tdm_load_tile_bf16((unsigned)(uintptr_t)&Bs[0][0][0],
                       Bt + (size_t)n0 * K, (unsigned)K, (unsigned)K, BN, BKg);
  }
#endif

  for (int it = 0; it < nk; ++it) {
    const int buf = it & 1;
    const int k0  = it * BKg;
#if USE_TDM
    if (wave == 0) {
      if (it + 1 < nk) {  // prefetch next tile into the other buffer
        tdm_load_tile_bf16((unsigned)(uintptr_t)&As[buf ^ 1][0][0],
                           A + (size_t)m0 * K + (k0 + BKg),
                           (unsigned)K, (unsigned)K, BM, BKg);
        tdm_load_tile_bf16((unsigned)(uintptr_t)&Bs[buf ^ 1][0][0],
                           Bt + (size_t)n0 * K + (k0 + BKg),
                           (unsigned)K, (unsigned)K, BN, BKg);
        __builtin_amdgcn_s_wait_tensorcnt(2);  // tile `it` complete
      } else {
        __builtin_amdgcn_s_wait_tensorcnt(0);
      }
    }
    __syncthreads();
#else
    {
      const int lrow = tid >> 1;
      const int lcol = (tid & 1) * 16;
      __syncthreads();
      const __bf16* ga = A  + (size_t)(m0 + lrow) * K + k0 + lcol;
      const __bf16* gb = Bt + (size_t)(n0 + lrow) * K + k0 + lcol;
#pragma unroll
      for (int i = 0; i < 16; ++i) As[buf][lrow][lcol + i] = ga[i];
#pragma unroll
      for (int i = 0; i < 16; ++i) Bs[buf][lrow][lcol + i] = gb[i];
      __syncthreads();
    }
#endif

    v16bf afr[2], bfr[4];
#pragma unroll
    for (int t = 0; t < 2; ++t) {  // A: lane=row, contiguous 16 k per half-wave
      const __bf16* src = &As[buf][wm * 32 + t * 16 + r][half * 16];
#pragma unroll
      for (int e = 0; e < 16; ++e) afr[t][e] = src[e];
    }
#pragma unroll
    for (int t = 0; t < 4; ++t) {  // B: lane=col, contiguous 16 k per half-wave
      const __bf16* src = &Bs[buf][wn * 64 + t * 16 + r][half * 16];
#pragma unroll
      for (int e = 0; e < 16; ++e) bfr[t][e] = src[e];
    }
#pragma unroll
    for (int i = 0; i < 2; ++i)
#pragma unroll
      for (int j = 0; j < 4; ++j)
        acc[i][j] = wmma_bf16(afr[i], bfr[j], acc[i][j]);

#if USE_TDM
    __syncthreads();  // compute done before buf is re-filled next-next iter
#endif
  }

#pragma unroll
  for (int i = 0; i < 2; ++i)
#pragma unroll
    for (int j = 0; j < 4; ++j)
#pragma unroll
      for (int e = 0; e < 8; ++e) {
        const int row = m0 + wm * 32 + i * 16 + e + half * 8;
        const int col = n0 + wn * 64 + j * 16 + r;
        float v = acc[i][j][e];
        if constexpr (EPI >= 1) v += bias[col];
        if constexpr (EPI == 1)
          Cb[(size_t)row * N + col] = (__bf16)gelu_f(v);
        else
          Cf[(size_t)row * N + col] = v;
      }
}

// ------------------------------- attention --------------------------------
// grid.x = S/128 (Q row block), grid.y = B*H. Head h reads proj column j*H+h,
// writes output column h*64+j (matches reference reshape/transpose quirk).
__global__ __launch_bounds__(256) void attention_wmma(
    const float* __restrict__ Q, const float* __restrict__ K,
    const float* __restrict__ V, float* __restrict__ O, float scale) {
  __shared__ __bf16 Qs[128][64 + PAD];
  __shared__ __bf16 Ks[64][64 + PAD];
  __shared__ __bf16 Vts[64][64 + PAD];   // Vt[dk][kv]
  __shared__ __bf16 Ps[8][16][64 + PAD]; // per-wave P tile

  const int tid  = threadIdx.x;
  const int lane = tid & 31;
  const int wave = tid >> 5;
  const int half = lane >> 4;
  const int r    = lane & 15;
  const int bh   = blockIdx.y;
  const int b    = bh >> 3, h = bh & 7;
  const int row0 = blockIdx.x * 128;
  const size_t base = (size_t)b * 512 * 512;

  {  // stage Q block (pre-scaled) as bf16
    int row = tid >> 1, j0 = (tid & 1) * 32;
    const float* src = Q + base + (size_t)(row0 + row) * 512 + h;
    for (int i = 0; i < 32; ++i)
      Qs[row][j0 + i] = (__bf16)(src[(size_t)(j0 + i) * 8] * scale);
  }

  const v8f vzero = {0.f, 0.f, 0.f, 0.f, 0.f, 0.f, 0.f, 0.f};
  v8f o[4];
#pragma unroll
  for (int j = 0; j < 4; ++j) o[j] = vzero;
  float rmax[8], rsum[8];
#pragma unroll
  for (int i = 0; i < 8; ++i) { rmax[i] = -1e30f; rsum[i] = 0.f; }

  for (int kb = 0; kb < 8; ++kb) {
    const int kv0 = kb * 64;
    __syncthreads();
    {  // stage K block row-major, V block transposed
      int krow = tid >> 2, jb = (tid & 3) * 16;
      const float* srcK = K + base + (size_t)(kv0 + krow) * 512 + h;
      for (int i = 0; i < 16; ++i) Ks[krow][jb + i] = (__bf16)srcK[(size_t)(jb + i) * 8];
      int vj = tid >> 2, kvb = (tid & 3) * 16;
      for (int i = 0; i < 16; ++i)
        Vts[vj][kvb + i] = (__bf16)V[base + (size_t)(kv0 + kvb + i) * 512 + vj * 8 + h];
    }
    __syncthreads();

    // S = Q_tile(16x64) @ K_blk^T  (k = dk = 64, two WMMA k-steps)
    v8f s[4];
#pragma unroll
    for (int j = 0; j < 4; ++j) s[j] = vzero;
#pragma unroll
    for (int kc = 0; kc < 2; ++kc) {
      v16bf a;
      const __bf16* qp = &Qs[wave * 16 + r][kc * 32 + half * 16];
#pragma unroll
      for (int e = 0; e < 16; ++e) a[e] = qp[e];
#pragma unroll
      for (int j = 0; j < 4; ++j) {
        v16bf bb;
        const __bf16* kp = &Ks[j * 16 + r][kc * 32 + half * 16];
#pragma unroll
        for (int e = 0; e < 16; ++e) bb[e] = kp[e];
        s[j] = wmma_bf16(a, bb, s[j]);
      }
    }

    // online softmax: row m = vgpr + 8*half lives in one 16-lane half-wave
#pragma unroll
    for (int i = 0; i < 8; ++i) {
      float bm = fmaxf(fmaxf(s[0][i], s[1][i]), fmaxf(s[2][i], s[3][i]));
      for (int off = 1; off < 16; off <<= 1) bm = fmaxf(bm, __shfl_xor(bm, off, 32));
      float nm = fmaxf(rmax[i], bm);
      float f  = __expf(rmax[i] - nm);
      rmax[i]  = nm;
      float bs = 0.f;
#pragma unroll
      for (int j = 0; j < 4; ++j) {
        float p = __expf(s[j][i] - nm);
        s[j][i] = p;
        bs += p;
      }
      for (int off = 1; off < 16; off <<= 1) bs += __shfl_xor(bs, off, 32);
      rsum[i] = rsum[i] * f + bs;
#pragma unroll
      for (int j = 0; j < 4; ++j) o[j][i] *= f;
    }

    // transpose P (C-layout -> A-layout) through per-wave LDS
#pragma unroll
    for (int j = 0; j < 4; ++j)
#pragma unroll
      for (int i = 0; i < 8; ++i)
        Ps[wave][i + half * 8][j * 16 + r] = (__bf16)s[j][i];
    __syncthreads();

    // O += P(16x64) @ V_blk(64x64)
#pragma unroll
    for (int kc = 0; kc < 2; ++kc) {
      v16bf a;
      const __bf16* pp = &Ps[wave][r][kc * 32 + half * 16];
#pragma unroll
      for (int e = 0; e < 16; ++e) a[e] = pp[e];
#pragma unroll
      for (int j = 0; j < 4; ++j) {
        v16bf bb;
        const __bf16* vp = &Vts[j * 16 + r][kc * 32 + half * 16];
#pragma unroll
        for (int e = 0; e < 16; ++e) bb[e] = vp[e];
        o[j] = wmma_bf16(a, bb, o[j]);
      }
    }
  }

#pragma unroll
  for (int j = 0; j < 4; ++j)
#pragma unroll
    for (int i = 0; i < 8; ++i) {
      const int row = row0 + wave * 16 + i + half * 8;
      const int col = h * 64 + j * 16 + r;
      O[base + (size_t)row * 512 + col] = o[j][i] / rsum[i];
    }
}

// --------------------------- residual + layernorm -------------------------
__global__ __launch_bounds__(256) void residual_ln(
    const float* __restrict__ X, const float* __restrict__ Dl,
    const float* __restrict__ G, const float* __restrict__ Bv,
    float* __restrict__ Y) {
  const int wave = threadIdx.x >> 5, lane = threadIdx.x & 31;
  const size_t row = (size_t)blockIdx.x * 8 + wave;
  const float* x = X  + row * 512;
  const float* d = Dl + row * 512;
  float t[16], sum = 0.f;
#pragma unroll
  for (int i = 0; i < 16; ++i) { t[i] = x[lane + i * 32] + d[lane + i * 32]; sum += t[i]; }
  for (int off = 1; off < 32; off <<= 1) sum += __shfl_xor(sum, off, 32);
  const float mean = sum * (1.0f / 512.0f);
  float vs = 0.f;
#pragma unroll
  for (int i = 0; i < 16; ++i) { float u = t[i] - mean; vs += u * u; }
  for (int off = 1; off < 32; off <<= 1) vs += __shfl_xor(vs, off, 32);
  const float rstd = rsqrtf(vs * (1.0f / 512.0f) + 1e-5f);
#pragma unroll
  for (int i = 0; i < 16; ++i) {
    const int c = lane + i * 32;
    Y[row * 512 + c] = G[c] * (t[i] - mean) * rstd + Bv[c];
  }
}

// --------------------------------- driver ---------------------------------

extern "C" void kernel_launch(void* const* d_in, const int* in_sizes, int n_in,
                              void* d_out, int out_size, void* d_ws, size_t ws_size,
                              hipStream_t stream) {
  (void)in_sizes; (void)n_in; (void)out_size; (void)ws_size;
  const float* x_in = (const float*)d_in[0];
  const float* WQ   = (const float*)d_in[1];
  const float* WK   = (const float*)d_in[2];
  const float* WV   = (const float*)d_in[3];
  const float* WO   = (const float*)d_in[4];
  const float* K1   = (const float*)d_in[5];
  const float* B1   = (const float*)d_in[6];
  const float* K2   = (const float*)d_in[7];
  const float* B2   = (const float*)d_in[8];
  const float* G1   = (const float*)d_in[9];
  const float* Bt1  = (const float*)d_in[10];
  const float* G2   = (const float*)d_in[11];
  const float* Bt2  = (const float*)d_in[12];

  const int Bb = 16, Ss = 512, Dd = 512, FFn = 2048, LL = 6, HH = 8;
  const int M = Bb * Ss;  // 8192

  char* p = (char*)d_ws;
  auto take = [&](size_t bytes) -> void* {
    char* q = p;
    p += (bytes + 255) & ~(size_t)255;
    return (void*)q;
  };
  // bf16 transposed weights (~38 MB) + activations (~143 MB); ws ~180 MB.
  __bf16* WQt = (__bf16*)take((size_t)LL * Dd * Dd * 2);
  __bf16* WKt = (__bf16*)take((size_t)LL * Dd * Dd * 2);
  __bf16* WVt = (__bf16*)take((size_t)LL * Dd * Dd * 2);
  __bf16* WOt = (__bf16*)take((size_t)LL * Dd * Dd * 2);
  __bf16* K1t = (__bf16*)take((size_t)LL * Dd * FFn * 2);
  __bf16* K2t = (__bf16*)take((size_t)LL * Dd * FFn * 2);
  float* xA    = (float*)take((size_t)M * Dd * 4);
  float* xB    = (float*)take((size_t)M * Dd * 4);
  float* Qf    = (float*)take((size_t)M * Dd * 4);  // also WO-out / FFN2-out
  float* Kf    = (float*)take((size_t)M * Dd * 4);
  float* Vf    = (float*)take((size_t)M * Dd * 4);
  float* attnf = (float*)take((size_t)M * Dd * 4);
  __bf16* xb   = (__bf16*)take((size_t)M * Dd * 2);
  __bf16* h1b  = (__bf16*)take((size_t)M * FFn * 2);

  // one-time weight transpose+convert
  dim3 tg((Dd * Dd + 255) / 256, LL);
  transpose_to_bf16_k<<<tg, 256, 0, stream>>>(WQ, WQt, Dd, Dd);
  transpose_to_bf16_k<<<tg, 256, 0, stream>>>(WK, WKt, Dd, Dd);
  transpose_to_bf16_k<<<tg, 256, 0, stream>>>(WV, WVt, Dd, Dd);
  transpose_to_bf16_k<<<tg, 256, 0, stream>>>(WO, WOt, Dd, Dd);
  dim3 tg1((Dd * FFn + 255) / 256, LL);
  transpose_to_bf16_k<<<tg1, 256, 0, stream>>>(K1, K1t, Dd, FFn);
  transpose_to_bf16_k<<<tg1, 256, 0, stream>>>(K2, K2t, FFn, Dd);

  (void)hipMemcpyAsync(xA, x_in, (size_t)M * Dd * 4, hipMemcpyDeviceToDevice, stream);

  const dim3 gqkv(M / BM, Dd / BN);   // (64, 4)
  const dim3 gff1(M / BM, FFn / BN);  // (64, 16)
  const dim3 gat(Ss / 128, Bb * HH);  // (4, 128)

  for (int l = 0; l < LL; ++l) {
    const __bf16* wq = WQt + (size_t)l * Dd * Dd;
    const __bf16* wk = WKt + (size_t)l * Dd * Dd;
    const __bf16* wv = WVt + (size_t)l * Dd * Dd;
    const __bf16* wo = WOt + (size_t)l * Dd * Dd;
    const __bf16* k1 = K1t + (size_t)l * Dd * FFn;
    const __bf16* k2 = K2t + (size_t)l * Dd * FFn;

    f32_to_bf16_k<<<2048, 256, 0, stream>>>(xA, xb, (long long)M * Dd);
    gemm_bf16_wmma<0><<<gqkv, 256, 0, stream>>>(xb, wq, nullptr, Qf, nullptr, M, Dd, Dd);
    gemm_bf16_wmma<0><<<gqkv, 256, 0, stream>>>(xb, wk, nullptr, Kf, nullptr, M, Dd, Dd);
    gemm_bf16_wmma<0><<<gqkv, 256, 0, stream>>>(xb, wv, nullptr, Vf, nullptr, M, Dd, Dd);

    attention_wmma<<<gat, 256, 0, stream>>>(Qf, Kf, Vf, attnf, 0.125f);

    f32_to_bf16_k<<<2048, 256, 0, stream>>>(attnf, xb, (long long)M * Dd);
    gemm_bf16_wmma<0><<<gqkv, 256, 0, stream>>>(xb, wo, nullptr, Qf, nullptr, M, Dd, Dd);
    residual_ln<<<M / 8, 256, 0, stream>>>(xA, Qf, G1 + (size_t)l * Dd,
                                           Bt1 + (size_t)l * Dd, xB);

    f32_to_bf16_k<<<2048, 256, 0, stream>>>(xB, xb, (long long)M * Dd);
    gemm_bf16_wmma<1><<<gff1, 256, 0, stream>>>(xb, k1, B1 + (size_t)l * FFn,
                                                nullptr, h1b, M, FFn, Dd);
    gemm_bf16_wmma<2><<<gqkv, 256, 0, stream>>>(h1b, k2, B2 + (size_t)l * Dd,
                                                Qf, nullptr, M, Dd, FFn);
    float* dst = (l == LL - 1) ? (float*)d_out : xA;
    residual_ln<<<M / 8, 256, 0, stream>>>(xB, Qf, G2 + (size_t)l * Dd,
                                           Bt2 + (size_t)l * Dd, dst);
  }
}
